// unit_agcn_77043123355730
// MI455X (gfx1250) — matrix operand order repaired and checked by reading, not compiled, not run on hardware
//
#include <hip/hip_runtime.h>
#include <hip/hip_bf16.h>

#define N_    16
#define C_IN  64
#define T_    512
#define V_    22
#define C_OUT 256
#define INTER 64
#define VP    32   // V padded to WMMA tile multiple
#define TF    4    // frames per workgroup

typedef __attribute__((ext_vector_type(16))) __bf16 v16bf;
typedef __attribute__((ext_vector_type(8)))  float  v8f;
typedef __attribute__((ext_vector_type(2)))  __bf16 bf16x2;

union FragU { uint4 q[2]; v16bf v; };

// A-operand fragment (16x32 bf16) from row-major tile, row stride rs (elements).
// Per CDNA5 ISA: lane<16 -> M=lane, K chunks {8h..8h+7} and {16+8h..16+8h+7}
// -> exactly two contiguous 16-byte loads per lane.
__device__ __forceinline__ v16bf ld_afrag(const __bf16* base, int row0, int col0,
                                          int rs, int lane) {
  const int half = (lane >> 4) & 1;
  const int m = lane & 15;
  const __bf16* p = base + (row0 + m) * rs + col0 + 8 * half;
  FragU f;
  f.q[0] = *(const uint4*)(p);
  f.q[1] = *(const uint4*)(p + 16);
  return f.v;
}

// B-operand fragment (32x16 bf16) from COLUMN-major storage: element (k,n) at
// base[n*rs + k]. Lane holds column n, K chunks {8h..8h+7} and {16+8h..16+8h+7}
// -> two 16-byte loads per lane.
__device__ __forceinline__ v16bf ld_bfrag(const __bf16* base, int row0, int col0,
                                          int rs, int lane) {
  const int half = (lane >> 4) & 1;
  const int n = lane & 15;
  const __bf16* p = base + (col0 + n) * rs + row0 + 8 * half;
  FragU f;
  f.q[0] = *(const uint4*)(p);
  f.q[1] = *(const uint4*)(p + 16);
  return f.v;
}

// ---------------- kernel 1: fp32 weights -> bf16 staging buffer ----------------
// ws layout (bf16): [0,4096) Wa ; [4096,8192) Wb ; [8192,24576) Wd
__global__ void agcn_cvt_weights(const float* __restrict__ Wa,
                                 const float* __restrict__ Wb,
                                 const float* __restrict__ Wd,
                                 __bf16* __restrict__ wsW) {
  const int i = blockIdx.x * blockDim.x + threadIdx.x;
  if (i >= 4096 + 4096 + 16384) return;
  float f;
  if (i < 4096)      f = Wa[i];
  else if (i < 8192) f = Wb[i - 4096];
  else               f = Wd[i - 8192];
  wsW[i] = (__bf16)f;
}

// ---------------- kernel 2: fully fused AGCN block ----------------
__global__ __launch_bounds__(128)
void agcn_fused(const float* __restrict__ x,
                const float* __restrict__ Ain,
                const float* __restrict__ ba,
                const float* __restrict__ bb,
                const float* __restrict__ bd,
                const __bf16* __restrict__ wsW,
                float* __restrict__ out) {
  __shared__ __align__(16) __bf16 smem[25600];   // 50 KB
  __bf16* WdB   = smem;             // [256][64] row-major (A operand, stage 6)
  __bf16* xb    = WdB + 16384;      // [64][32]  row-major (A operand, stage 5)
  __bf16* xbT   = xb + 2048;        // [32][64]  x^T = col-major B (stage 2)
  __bf16* A1t   = xbT + 2048;       // [32][64]  A1^T row-major A (stage 3) | alias scoresT f32[32][32]
  __bf16* A2t   = A1t + 2048;       // [32][64]  A2 col-major B (stage 3)   | alias zbT
  __bf16* Acomb = A2t + 2048;       // [32][32]  row-major [u][w] = col-major B (stage 5)
  float*  scoresT = (float*)A1t;    // scoresT[v][u] = scores[u][v]
  __bf16* zbT     = A2t;            // [32][64]  z^T [u][c] = col-major B (stage 6)

  const int tid  = threadIdx.x;
  const int lane = tid & 31;
  const int w    = tid >> 5;
  const int half = (lane >> 4) & 1;
  const int ln   = lane & 15;

  // stage 0: Wd -> LDS (32 KB = 2048 x uint4)
  {
    const uint4* src = (const uint4*)(wsW + 8192);
    uint4* dst = (uint4*)WdB;
    for (int i = tid; i < 2048; i += 128) dst[i] = src[i];
  }
  __syncthreads();

  for (int f = 0; f < TF; ++f) {
    const int frame = blockIdx.x * TF + f;
    const int n = frame / T_;
    const int t = frame % T_;

    // stage 1: x[n,:,t,:] -> xb [c][v] and xbT [v][c] (bf16, zero-padded)
    for (int i = tid; i < C_IN * VP; i += 128) {
      const int c = i >> 5, v = i & 31;
      const float val = (v < V_) ? x[((n * C_IN + c) * T_ + t) * V_ + v] : 0.0f;
      const __bf16 h = (__bf16)val;
      xb[c * VP + v]  = h;
      xbT[v * 64 + c] = h;
    }
    __syncthreads();

    // prefetch next frame's x slice (64 rows x 88 B) and A slice (1936 B)
    // into cache under this frame's compute (global_prefetch_b8: no counter,
    // no VGPR return; divergence-safe outside WMMA regions)
    if (f + 1 < TF) {
      const int nf = frame + 1;
      const int n2 = nf / T_, t2 = nf % T_;
      {
        const int c = tid & 63;
        const char* p = (const char*)(x + ((n2 * C_IN + c) * T_ + t2) * V_)
                        + (tid >> 6) * 64;
        __builtin_prefetch(p, 0, 1);
      }
      if (tid < 31) {
        const char* p = (const char*)(Ain + ((size_t)(n2 * T_ + t2) * V_) * V_)
                        + tid * 64;
        __builtin_prefetch(p, 0, 1);
      }
    }

    // stage 2: A1 = Wa@x + ba ; A2 = Wb@x + bb  -> stored TRANSPOSED [v][i]
    // (16 tiles of 16x16, 4 per wave; weight A-frags stream from L2)
#pragma unroll
    for (int j = 0; j < 4; ++j) {
      const int tile = w + 4 * j;             // 0..15
      const bool second = (tile >= 8);
      const int mt = (tile & 7) >> 1;
      const int nt = tile & 1;
      const __bf16* Wsrc = second ? (wsW + 4096) : wsW;
      const float* bias  = second ? bb : ba;
      __bf16* dst        = second ? A2t : A1t;
      v8f acc = {};
#pragma unroll
      for (int ks = 0; ks < 2; ++ks) {
        v16bf a = ld_afrag(Wsrc, mt * 16, ks * 32, C_IN, lane);
        v16bf b = ld_bfrag(xbT, ks * 32, nt * 16, 64, lane);
        acc = __builtin_amdgcn_wmma_f32_16x16x32_bf16(false, a, false, b, (short)0,
                                                      acc, false, false);
      }
      const int col = nt * 16 + ln;           // v index
#pragma unroll
      for (int i = 0; i < 8; i += 2) {        // paired rows -> packed store
        const int row = mt * 16 + i + 8 * half;  // inter-channel index
        bf16x2 pk = { (__bf16)(acc[i] + bias[row]),
                      (__bf16)(acc[i + 1] + bias[row + 1]) };
        *(bf16x2*)(dst + col * 64 + row) = pk;
      }
    }
    __syncthreads();

    // stage 3: scores = (A1^T @ A2) / INTER -> scoresT[v][u] f32 (aliases A1t)
    {
      const int mt = w >> 1, nt = w & 1;      // 2x2 tiles, one per wave
      v8f acc = {};
#pragma unroll
      for (int ks = 0; ks < 2; ++ks) {
        v16bf a = ld_afrag(A1t, mt * 16, ks * 32, 64, lane);   // A1^T rows = u
        v16bf b = ld_bfrag(A2t, ks * 32, nt * 16, 64, lane);   // cols = v
        acc = __builtin_amdgcn_wmma_f32_16x16x32_bf16(false, a, false, b, (short)0,
                                                      acc, false, false);
      }
      __syncthreads();                        // all A1t reads done before aliased write
      const int col = nt * 16 + ln;           // v
#pragma unroll
      for (int i = 0; i < 8; i += 2) {
        const int row = mt * 16 + i + 8 * half;  // u
        float2 pr = make_float2(acc[i] * (1.0f / INTER), acc[i + 1] * (1.0f / INTER));
        *(float2*)(scoresT + col * VP + row) = pr;
      }
    }
    __syncthreads();

    // stage 4: softmax over u per column v (row-contiguous in scoresT);
    //          Acomb[u][w] = A + 1e-6 + softmax, zero-padded to 32x32
    if (tid < VP) {
      const int v = tid;
      if (v < V_) {
        const float* srow = scoresT + v * VP;
        float mx = -1e30f;
        for (int u = 0; u < V_; ++u) mx = fmaxf(mx, srow[u]);
        float e[V_];
        float s = 0.0f;
        for (int u = 0; u < V_; ++u) { e[u] = __expf(srow[u] - mx); s += e[u]; }
        const float inv = 1.0f / s;
        const float* Acol = Ain + ((size_t)(n * T_ + t) * V_) * V_ + v;
        for (int u = 0; u < V_; ++u)
          Acomb[u * VP + v] = (__bf16)(Acol[u * V_] + 1e-6f + e[u] * inv);
        for (int u = V_; u < VP; ++u)
          Acomb[u * VP + v] = (__bf16)0.0f;   // zero pad rows u>=22
      } else {
        for (int u = 0; u < VP; ++u)
          Acomb[u * VP + v] = (__bf16)0.0f;   // zero pad cols w>=22 (K padding!)
      }
    }
    __syncthreads();

    // stage 5: Z[c][u] = sum_w x[c][w]*Acomb[u][w] -> stored TRANSPOSED zbT[u][c]
#pragma unroll
    for (int j = 0; j < 2; ++j) {
      const int tile = w + 4 * j;             // 0..7
      const int mt = tile >> 1, nt = tile & 1;
      v16bf a = ld_afrag(xb, mt * 16, 0, VP, lane);
      v16bf b = ld_bfrag(Acomb, 0, nt * 16, VP, lane);
      v8f acc = {};
      acc = __builtin_amdgcn_wmma_f32_16x16x32_bf16(false, a, false, b, (short)0,
                                                    acc, false, false);
      const int col = nt * 16 + ln;           // u
#pragma unroll
      for (int i = 0; i < 8; i += 2) {
        const int row = mt * 16 + i + 8 * half;  // c
        bf16x2 pk = { (__bf16)acc[i], (__bf16)acc[i + 1] };
        *(bf16x2*)(zbT + col * 64 + row) = pk;   // aliases A2t (reads done stage 3)
      }
    }
    __syncthreads();

    // stage 6: out = relu(Wd @ Z + bd)  (32 tiles, 8 per wave, K=64)
#pragma unroll
    for (int j = 0; j < 8; ++j) {
      const int tile = w + 4 * j;             // 0..31
      const int mt = tile >> 1, nt = tile & 1;
      v8f acc = {};
#pragma unroll
      for (int ks = 0; ks < 2; ++ks) {
        v16bf a = ld_afrag(WdB, mt * 16, ks * 32, C_IN, lane);
        v16bf b = ld_bfrag(zbT, ks * 32, nt * 16, 64, lane);
        acc = __builtin_amdgcn_wmma_f32_16x16x32_bf16(false, a, false, b, (short)0,
                                                      acc, false, false);
      }
      const int col = nt * 16 + ln;           // v
      if (col < V_) {                         // divergent only around stores
#pragma unroll
        for (int i = 0; i < 8; ++i) {
          const int o = mt * 16 + i + 8 * half;
          out[((n * C_OUT + o) * T_ + t) * V_ + col] = fmaxf(acc[i] + bd[o], 0.0f);
        }
      }
    }
    __syncthreads();                          // protect LDS before next frame
  }
}

extern "C" void kernel_launch(void* const* d_in, const int* in_sizes, int n_in,
                              void* d_out, int out_size, void* d_ws, size_t ws_size,
                              hipStream_t stream) {
  const float* x  = (const float*)d_in[0];
  const float* A  = (const float*)d_in[1];
  const float* Wa = (const float*)d_in[2];
  const float* ba = (const float*)d_in[3];
  const float* Wb = (const float*)d_in[4];
  const float* bb = (const float*)d_in[5];
  const float* Wd = (const float*)d_in[6];
  const float* bd = (const float*)d_in[7];
  float* out = (float*)d_out;
  __bf16* wsW = (__bf16*)d_ws;

  // stage 1: fp32 -> bf16 weight staging (24576 elements)
  agcn_cvt_weights<<<(24576 + 255) / 256, 256, 0, stream>>>(Wa, Wb, Wd, wsW);

  // stage 2: fused AGCN, 4 frames per 128-thread (4-wave) workgroup
  const int frames = N_ * T_;  // 8192
  agcn_fused<<<frames / TF, 128, 0, stream>>>(x, A, ba, bb, bd, wsW, out);
}
